// Classifier_61838939127881
// MI455X (gfx1250) — matrix operand order repaired and checked by reading, not compiled, not run on hardware
//
#include <hip/hip_runtime.h>

// CDNA5 / gfx1250: wave32, WMMA 16x16x32 bf16 with f32 accumulate.

typedef __attribute__((ext_vector_type(16))) __bf16 v16bf;
typedef __attribute__((ext_vector_type(8)))  float  v8f;

#define K_DIM      512
#define N_DIM      2000          // NUM_CLASSES * 2
#define M_DIM      16384
#define KSTEPS     16            // 512 / 32
#define NT_PER_WG  4             // 4 n-tiles of 16 -> 64 columns per workgroup
#define M_PER_WG   512           // 8 waves * 4 m-tiles * 16 rows
#define A_SLOTS    ((M_DIM / 16) * KSTEPS * 32)   // 524288 fragments-slots
#define XF_BYTES   ((size_t)M_DIM * K_DIM * 2)    // 16.78 MB bf16 workspace

// ---------------------------------------------------------------------------
// Fragment helpers
// ---------------------------------------------------------------------------
__device__ __forceinline__ v16bf cvt_frag(float4 a0, float4 a1, float4 a2, float4 a3)
{
    const float av[16] = { a0.x,a0.y,a0.z,a0.w, a1.x,a1.y,a1.z,a1.w,
                           a2.x,a2.y,a2.z,a2.w, a3.x,a3.y,a3.z,a3.w };
    v16bf f;
    #pragma unroll
    for (int j = 0; j < 16; ++j) f[j] = (__bf16)av[j];
    return f;
}

__device__ __forceinline__ v16bf load_a_frag_f32(const float* __restrict__ xrow,
                                                 int kb, int koffA)
{
    // A fragment (16x32 bf16): elems 0..7 = K kb+koffA+0..7 (V0-V3),
    //                          elems 8..15 = K kb+16+koffA+0..7 (V4-V7)
    const float4* pa = (const float4*)(xrow + kb + koffA);
    const float4* pb = (const float4*)(xrow + kb + 16 + koffA);
    return cvt_frag(pa[0], pa[1], pb[0], pb[1]);
}

__device__ __forceinline__ void lse_store(float v, int n, size_t m, float* __restrict__ Out)
{
    // Logit pairs (t=0,1) are adjacent columns -> adjacent lanes (XOR 1).
    float vp = __shfl_xor(v, 1, 32);
    float mx = fmaxf(v, vp);
    float lse = mx + __logf(__expf(v - mx) + __expf(vp - mx));
    if (n < N_DIM) Out[m * N_DIM + n] = v - lse;
}

// ---------------------------------------------------------------------------
// Pass 1 (fast path): convert X f32 -> bf16, pre-swizzled to A-fragment order.
// Slot s = (mtile*KSTEPS + ks)*32 + lane ; each slot = 32 B (16 bf16).
// ---------------------------------------------------------------------------
__global__ __launch_bounds__(256)
void convert_x_kernel(const float* __restrict__ X, v16bf* __restrict__ Xf)
{
    const int s  = blockIdx.x * 256 + threadIdx.x;
    const int sl = s & 31;
    const int ks = (s >> 5) & (KSTEPS - 1);
    const int mt = s >> 9;
    const int m  = mt * 16 + (sl & 15);
    const int k0 = ks * 32 + ((sl < 16) ? 0 : 8);
    const float4* p0 = (const float4*)(X + (size_t)m * K_DIM + k0);
    const float4* p1 = (const float4*)(X + (size_t)m * K_DIM + k0 + 16);
    Xf[s] = cvt_frag(p0[0], p0[1], p1[0], p1[1]);
}

// ---------------------------------------------------------------------------
// Shared W staging: f32 -> bf16 into LDS, B-fragment layout.
// B fragment (32x16, bf16): lane L holds column n = L&15,
// K = kb + (L<16 ? 0 : 8) + {0..7, 16..23}  (16 bf16 = 32 B / lane)
// ---------------------------------------------------------------------------
__device__ __forceinline__ void stage_w(const float* __restrict__ W,
                                        v16bf* ldsW, int tid, int nBase)
{
    for (int s = tid; s < NT_PER_WG * KSTEPS * 32; s += 256) {
        const int sl = s & 31;
        const int ks = (s >> 5) & (KSTEPS - 1);
        const int nt = s >> 9;
        const int n  = nBase + nt * 16 + (sl & 15);
        const int k0 = ks * 32 + ((sl < 16) ? 0 : 8);
        v16bf frag;
        if (n < N_DIM) {
            const float4* p = (const float4*)(W + (size_t)n * K_DIM + k0);
            frag = cvt_frag(p[0], p[1], p[4], p[5]);
        } else {
            #pragma unroll
            for (int j = 0; j < 16; ++j) frag[j] = (__bf16)0.0f;
        }
        ldsW[s] = frag;
    }
}

// ---------------------------------------------------------------------------
// Fast-path GEMM: A fragments read pre-converted/pre-swizzled from workspace.
// Inner loop per K-step: 4 vmem b128 + 8 ds_load_b128 + 8 independent WMMAs.
// ---------------------------------------------------------------------------
__global__ __launch_bounds__(256)
void classifier_wmma_pre(const v16bf* __restrict__ Xf,
                         const float* __restrict__ W,
                         const float* __restrict__ Bias,
                         float* __restrict__ Out)
{
    __shared__ v16bf ldsW[NT_PER_WG * KSTEPS * 32];   // 64 KB
    const int tid   = threadIdx.x;
    const int lane  = tid & 31;
    const int wave  = tid >> 5;
    const int nBase = blockIdx.x * (NT_PER_WG * 16);

    stage_w(W, ldsW, tid, nBase);
    __syncthreads();

    const int nl   = lane & 15;
    const int mOff = (lane < 16) ? 0 : 8;   // C/D layout: lanes>=16 hold M+8

    #pragma unroll 1
    for (int mtp = 0; mtp < 2; ++mtp) {
        const int m0 = blockIdx.y * M_PER_WG + wave * 64 + mtp * 32;
        // A-fragment slots for m-tiles m0/16 and m0/16+1
        const v16bf* xf0 = Xf + (size_t)(m0 >> 4) * (KSTEPS * 32) + lane;
        const v16bf* xf1 = xf0 + KSTEPS * 32;

        v8f acc0[NT_PER_WG], acc1[NT_PER_WG];
        #pragma unroll
        for (int nt = 0; nt < NT_PER_WG; ++nt)
            #pragma unroll
            for (int i = 0; i < 8; ++i) { acc0[nt][i] = 0.0f; acc1[nt][i] = 0.0f; }

        #pragma unroll 2
        for (int ks = 0; ks < KSTEPS; ++ks) {
            v16bf afrag0 = xf0[ks * 32];
            v16bf afrag1 = xf1[ks * 32];
            v16bf bfrag[NT_PER_WG];
            #pragma unroll
            for (int nt = 0; nt < NT_PER_WG; ++nt)
                bfrag[nt] = ldsW[(nt * KSTEPS + ks) * 32 + lane];
            #pragma unroll
            for (int nt = 0; nt < NT_PER_WG; ++nt) {
                acc0[nt] = __builtin_amdgcn_wmma_f32_16x16x32_bf16(
                    false, afrag0, false, bfrag[nt], (short)0, acc0[nt], false, false);
                acc1[nt] = __builtin_amdgcn_wmma_f32_16x16x32_bf16(
                    false, afrag1, false, bfrag[nt], (short)0, acc1[nt], false, false);
            }
        }

        #pragma unroll
        for (int nt = 0; nt < NT_PER_WG; ++nt) {
            const int n = nBase + nt * 16 + nl;
            const float bias = (n < N_DIM) ? Bias[n] : 0.0f;
            #pragma unroll
            for (int i = 0; i < 8; ++i) {
                lse_store(acc0[nt][i] + bias, n, (size_t)(m0 + mOff + i), Out);
                lse_store(acc1[nt][i] + bias, n, (size_t)(m0 + 16 + mOff + i), Out);
            }
        }
    }
}

// ---------------------------------------------------------------------------
// Fallback GEMM (no workspace): convert A f32->bf16 in the K-loop.
// ---------------------------------------------------------------------------
__global__ __launch_bounds__(256)
void classifier_wmma_fused(const float* __restrict__ X,
                           const float* __restrict__ W,
                           const float* __restrict__ Bias,
                           float* __restrict__ Out)
{
    __shared__ v16bf ldsW[NT_PER_WG * KSTEPS * 32];   // 64 KB
    const int tid   = threadIdx.x;
    const int lane  = tid & 31;
    const int wave  = tid >> 5;
    const int nBase = blockIdx.x * (NT_PER_WG * 16);

    stage_w(W, ldsW, tid, nBase);
    __syncthreads();

    const int nl    = lane & 15;
    const int mOff  = (lane < 16) ? 0 : 8;
    const int koffA = (lane < 16) ? 0 : 8;

    #pragma unroll 1
    for (int mtp = 0; mtp < 2; ++mtp) {
        const int m0 = blockIdx.y * M_PER_WG + wave * 64 + mtp * 32;
        const float* xrow0 = X + (size_t)(m0 + nl) * K_DIM;
        const float* xrow1 = xrow0 + (size_t)16 * K_DIM;

        v8f acc0[NT_PER_WG], acc1[NT_PER_WG];
        #pragma unroll
        for (int nt = 0; nt < NT_PER_WG; ++nt)
            #pragma unroll
            for (int i = 0; i < 8; ++i) { acc0[nt][i] = 0.0f; acc1[nt][i] = 0.0f; }

        #pragma unroll 2
        for (int ks = 0; ks < KSTEPS; ++ks) {
            const int kb = ks * 32;
            v16bf afrag0 = load_a_frag_f32(xrow0, kb, koffA);
            v16bf afrag1 = load_a_frag_f32(xrow1, kb, koffA);
            v16bf bfrag[NT_PER_WG];
            #pragma unroll
            for (int nt = 0; nt < NT_PER_WG; ++nt)
                bfrag[nt] = ldsW[(nt * KSTEPS + ks) * 32 + lane];
            #pragma unroll
            for (int nt = 0; nt < NT_PER_WG; ++nt) {
                acc0[nt] = __builtin_amdgcn_wmma_f32_16x16x32_bf16(
                    false, afrag0, false, bfrag[nt], (short)0, acc0[nt], false, false);
                acc1[nt] = __builtin_amdgcn_wmma_f32_16x16x32_bf16(
                    false, afrag1, false, bfrag[nt], (short)0, acc1[nt], false, false);
            }
        }

        #pragma unroll
        for (int nt = 0; nt < NT_PER_WG; ++nt) {
            const int n = nBase + nt * 16 + nl;
            const float bias = (n < N_DIM) ? Bias[n] : 0.0f;
            #pragma unroll
            for (int i = 0; i < 8; ++i) {
                lse_store(acc0[nt][i] + bias, n, (size_t)(m0 + mOff + i), Out);
                lse_store(acc1[nt][i] + bias, n, (size_t)(m0 + 16 + mOff + i), Out);
            }
        }
    }
}

extern "C" void kernel_launch(void* const* d_in, const int* in_sizes, int n_in,
                              void* d_out, int out_size, void* d_ws, size_t ws_size,
                              hipStream_t stream) {
    (void)in_sizes; (void)n_in; (void)out_size;
    const float* X    = (const float*)d_in[0];   // [16384, 512]
    const float* W    = (const float*)d_in[1];   // [1000, 2, 512] == [2000, 512]
    const float* Bias = (const float*)d_in[2];   // [1000, 2] == [2000]
    float* Out = (float*)d_out;                  // [16384, 1000, 2] == [16384, 2000]

    dim3 grid((N_DIM + NT_PER_WG * 16 - 1) / (NT_PER_WG * 16),   // 32 n-blocks
              M_DIM / M_PER_WG);                                  // 32 m-chunks

    if (ws_size >= XF_BYTES) {
        v16bf* Xf = (v16bf*)d_ws;
        convert_x_kernel<<<A_SLOTS / 256, 256, 0, stream>>>(X, Xf);
        classifier_wmma_pre<<<grid, 256, 0, stream>>>(Xf, W, Bias, Out);
    } else {
        classifier_wmma_fused<<<grid, 256, 0, stream>>>(X, W, Bias, Out);
    }
}
